// SelfAttentionWithCache_61229053772468
// MI455X (gfx1250) — matrix operand order repaired and checked by reading, not compiled, not run on hardware
//
#include <hip/hip_runtime.h>
#include <hip/hip_bf16.h>
#include <math.h>

#define BSZ 4
#define TGT 512
#define KVL 4096
#define DIM 2048
#define NH  16
#define HD  128
#define KVT 64

typedef __attribute__((ext_vector_type(16))) __bf16 v16bf;
typedef __attribute__((ext_vector_type(8)))  __bf16 v8bf;
typedef __attribute__((ext_vector_type(4)))  __bf16 v4bf;
typedef __attribute__((ext_vector_type(8)))  float  v8f;
typedef __attribute__((ext_vector_type(4)))  float  v4f;

// Combine two 8-element halves into a 16-element bf16 WMMA fragment.
__device__ __forceinline__ v16bf make_frag(const __bf16* lo, const __bf16* hi) {
    v8bf l = *(const v8bf*)lo;
    v8bf h = *(const v8bf*)hi;
    v16bf r;
#pragma unroll
    for (int i = 0; i < 8; ++i) { r[i] = l[i]; r[i + 8] = h[i]; }
    return r;
}

// A-matrix fragment (16x32 bf16, ISA 7.12.2): lane 0-15 rows; half-wave selects
// K offset +0 / +8 within each 16-wide K group. Tile row-major [row][K].
template <int RS>
__device__ __forceinline__ v16bf load_a_frag(const __bf16* tile, int row, int kstep, int lane) {
    int base = (lane >> 4) << 3;  // 0 or 8
    const __bf16* p = tile + row * RS + kstep;
    return make_frag(p + base, p + 16 + base);
}

// B-matrix fragment (32x16 bf16): lane&15 = N column, half-wave selects K offset +0 / +8 within each 16-wide K group.
// Tile stores B^T rows: row n holds the 32 K values contiguously.
template <int RS>
__device__ __forceinline__ v16bf load_b_frag(const __bf16* tile, int nrow, int kstep, int lane) {
    int kb = ((lane >> 4) << 3) + kstep;  // 0 or 8, + kstep
    const __bf16* p = tile + nrow * RS + kb;
    return make_frag(p, p + 16);
}

__device__ __forceinline__ v8f wmma_bf16(v16bf a, v16bf b, v8f c) {
    return __builtin_amdgcn_wmma_f32_16x16x32_bf16(false, a, false, b, (short)0, c, false, false);
}

// ---------------------------------------------------------------------------
// Bulk copy of the unchanged (first KV-TGT) cache rows, float4-vectorized.
// All strides constexpr: batch stride is 2^21 float4 -> pure shifts, no divs.
// ---------------------------------------------------------------------------
__global__ __launch_bounds__(256)
void copy_old_cache_kernel(const float* __restrict__ src, float* __restrict__ dst) {
    constexpr size_t N4PB = (size_t)(KVL - TGT) * DIM / 4;  // per-batch float4 count
    constexpr size_t BS4  = (size_t)KVL * DIM / 4;          // batch stride (2^21)
    constexpr size_t TOTAL = (size_t)BSZ * N4PB;
    size_t step = (size_t)gridDim.x * 256;
    for (size_t i = (size_t)blockIdx.x * 256 + threadIdx.x; i < TOTAL; i += step) {
        size_t bb  = i / N4PB;
        size_t rr  = i - bb * N4PB;
        size_t off = bb * BS4 + rr;
        ((v4f*)dst)[off] = ((const v4f*)src)[off];
    }
}

// ---------------------------------------------------------------------------
// GEMM: out = (A[M,K] @ W[N,K]^T + bias) * alpha, bf16 WMMA, f32 accumulate.
// Output rows remapped with compile-time seg=TGT (power of 2):
//   r -> (r>>9)*seg_stride + row_off + (r&511)     (KV-cache tail writes)
// WG = 256 threads (8 wave32), tile 128x128, K-step 32, double-buffered LDS:
// global loads for step k+1 issued before computing step k, one barrier/step,
// global_prefetch_b8 hints for step k+2.
// ---------------------------------------------------------------------------
__global__ __launch_bounds__(256)
void gemm_bias_wmma_kernel(const float* __restrict__ A, const float* __restrict__ W,
                           const float* __restrict__ bias, float* __restrict__ out,
                           float alpha, int K, int N, int seg_stride, int row_off) {
    __shared__ __attribute__((aligned(16))) __bf16 As[2][128 * 32];
    __shared__ __attribute__((aligned(16))) __bf16 Bs[2][128 * 32];

    const int tid  = threadIdx.x;
    const int lane = tid & 31;
    const int wid  = tid >> 5;
    const int mtile = blockIdx.y * 128;
    const int ntile = blockIdx.x * 128;
    const int wm = (wid & 3) * 32;   // wave covers 32 rows
    const int wn = (wid >> 2) * 64;  // wave covers 64 cols
    const int halfsel = lane >> 4;
    const int ncol    = lane & 15;

    // Per-thread staging slots: thread t owns float4 f = t + 256*i of each tile.
    int rr[4], cc[4];
#pragma unroll
    for (int i = 0; i < 4; ++i) {
        int f = tid + 256 * i;
        rr[i] = f >> 3;         // row within 128
        cc[i] = (f & 7) * 4;    // K column within 32
    }

    v8f acc[2][4];
#pragma unroll
    for (int mi = 0; mi < 2; ++mi)
#pragma unroll
        for (int ni = 0; ni < 4; ++ni) acc[mi][ni] = (v8f)0.0f;

    v4f a4[4], w4[4];
    const int NK = K >> 5;

    // prologue: fetch + stage tile 0
#pragma unroll
    for (int i = 0; i < 4; ++i) {
        a4[i] = *(const v4f*)(A + (size_t)(mtile + rr[i]) * K + cc[i]);
        w4[i] = *(const v4f*)(W + (size_t)(ntile + rr[i]) * K + cc[i]);
    }
#pragma unroll
    for (int i = 0; i < 4; ++i) {
        v4bf ab, wb;
#pragma unroll
        for (int j = 0; j < 4; ++j) { ab[j] = (__bf16)a4[i][j]; wb[j] = (__bf16)w4[i][j]; }
        *(v4bf*)(&As[0][rr[i] * 32 + cc[i]]) = ab;
        *(v4bf*)(&Bs[0][rr[i] * 32 + cc[i]]) = wb;
    }
    __syncthreads();

    for (int kk = 0; kk < NK; ++kk) {
        const int cur = kk & 1;
        const bool more = (kk + 1) < NK;

        if (more) {  // issue next-tile global loads before computing
            int k1 = (kk + 1) << 5;
#pragma unroll
            for (int i = 0; i < 4; ++i) {
                a4[i] = *(const v4f*)(A + (size_t)(mtile + rr[i]) * K + k1 + cc[i]);
                w4[i] = *(const v4f*)(W + (size_t)(ntile + rr[i]) * K + k1 + cc[i]);
            }
        }
        if (kk + 2 < NK) {  // prefetch hints for tile k+2 (global_prefetch_b8)
            int k2 = (kk + 2) << 5;
#pragma unroll
            for (int i = 0; i < 4; i += 2) {
                __builtin_prefetch(A + (size_t)(mtile + rr[i]) * K + k2 + cc[i], 0, 3);
                __builtin_prefetch(W + (size_t)(ntile + rr[i]) * K + k2 + cc[i], 0, 3);
            }
        }

        // compute from current buffer
        v16bf af[2], bf[4];
#pragma unroll
        for (int mi = 0; mi < 2; ++mi)
            af[mi] = load_a_frag<32>(&As[cur][0], wm + mi * 16 + ncol, 0, lane);
#pragma unroll
        for (int ni = 0; ni < 4; ++ni)
            bf[ni] = load_b_frag<32>(&Bs[cur][0], wn + ni * 16 + ncol, 0, lane);
#pragma unroll
        for (int mi = 0; mi < 2; ++mi)
#pragma unroll
            for (int ni = 0; ni < 4; ++ni)
                acc[mi][ni] = wmma_bf16(af[mi], bf[ni], acc[mi][ni]);

        if (more) {  // stage next tile into the other buffer
            const int nxt = cur ^ 1;
#pragma unroll
            for (int i = 0; i < 4; ++i) {
                v4bf ab, wb;
#pragma unroll
                for (int j = 0; j < 4; ++j) { ab[j] = (__bf16)a4[i][j]; wb[j] = (__bf16)w4[i][j]; }
                *(v4bf*)(&As[nxt][rr[i] * 32 + cc[i]]) = ab;
                *(v4bf*)(&Bs[nxt][rr[i] * 32 + cc[i]]) = wb;
            }
        }
        __syncthreads();
    }

    // Epilogue: C layout: VGPR j holds M=j (lanes 0-15) / M=j+8 (lanes 16-31).
#pragma unroll
    for (int mi = 0; mi < 2; ++mi)
#pragma unroll
        for (int ni = 0; ni < 4; ++ni) {
            int n = ntile + wn + ni * 16 + ncol;
            float bn = bias[n];
#pragma unroll
            for (int j = 0; j < 8; ++j) {
                int m = mtile + wm + mi * 16 + j + halfsel * 8;
                float v = (acc[mi][ni][j] + bn) * alpha;
                int orow = (m >> 9) * seg_stride + row_off + (m & (TGT - 1));
                out[(size_t)orow * N + n] = v;
            }
        }
}

// ---------------------------------------------------------------------------
// Flash attention: one WG = 128 q-rows of one (b,h); 8 waves x 16 rows each.
// KV streamed in tiles of 64 through LDS (bf16), online softmax, WMMA for
// both QK^T and PV. O accumulated in f32 registers (16x128 per wave).
// ---------------------------------------------------------------------------
__global__ __launch_bounds__(256)
void flash_attn_wmma_kernel(const float* __restrict__ q, const float* __restrict__ kcache,
                            const float* __restrict__ vcache, const float* __restrict__ mask,
                            float* __restrict__ attn_out) {
    __shared__ __attribute__((aligned(16))) __bf16 Ks[KVT * HD];    // [kv][d]   16KB
    __shared__ __attribute__((aligned(16))) __bf16 VTs[HD * KVT];   // [d][kv]   16KB
    __shared__ __attribute__((aligned(16))) __bf16 Ps[8][16 * KVT]; // per-wave  16KB

    const int tid  = threadIdx.x;
    const int lane = tid & 31;
    const int wid  = tid >> 5;
    const int b = blockIdx.z;
    const int h = blockIdx.y;
    const int qtile = blockIdx.x * 128;
    const int halfsel = lane >> 4;
    const int ncol    = lane & 15;

    // Q fragments for this wave's 16 rows: f32 -> bf16, kept resident (4 x v16bf).
    v16bf aq[4];
    {
        int m = qtile + wid * 16 + ncol;
        const float* qp = q + ((size_t)(b * TGT + m)) * DIM + h * HD;
        int base = halfsel * 8;
#pragma unroll
        for (int s = 0; s < 4; ++s) {
            v16bf f;
#pragma unroll
            for (int i = 0; i < 8; ++i) f[i]     = (__bf16)qp[32 * s + base + i];
#pragma unroll
            for (int i = 0; i < 8; ++i) f[8 + i] = (__bf16)qp[32 * s + 16 + base + i];
            aq[s] = f;
        }
    }

    v8f o[8];
#pragma unroll
    for (int t = 0; t < 8; ++t) o[t] = (v8f)0.0f;
    float mrow[8], lrow[8];
#pragma unroll
    for (int j = 0; j < 8; ++j) { mrow[j] = -1e30f; lrow[j] = 0.0f; }

    const float* kb = kcache + (size_t)b * KVL * DIM + h * HD;
    const float* vb = vcache + (size_t)b * KVL * DIM + h * HD;
    __bf16* pw = &Ps[wid][0];

    for (int kt = 0; kt < KVL; kt += KVT) {
        // Stage K tile [64][128] and transposed V tile [128][64] as bf16.
#pragma unroll
        for (int i = 0; i < 8; ++i) {
            int f  = tid + 256 * i;        // float4 index, 32 per row
            int r  = f >> 5;
            int c4 = (f & 31) * 4;
            v4f k4 = *(const v4f*)(kb + (size_t)(kt + r) * DIM + c4);
            v4f v4 = *(const v4f*)(vb + (size_t)(kt + r) * DIM + c4);
            v4bf kbf;
#pragma unroll
            for (int j = 0; j < 4; ++j) kbf[j] = (__bf16)k4[j];
            *(v4bf*)(Ks + r * HD + c4) = kbf;
#pragma unroll
            for (int j = 0; j < 4; ++j) VTs[(c4 + j) * KVT + r] = (__bf16)v4[j];
        }
        // Prefetch hints for next KV tile (global_prefetch_b8).
        if (kt + KVT < KVL) {
#pragma unroll
            for (int i = 0; i < 8; i += 2) {
                int f  = tid + 256 * i;
                int r  = f >> 5;
                int c4 = (f & 31) * 4;
                __builtin_prefetch(kb + (size_t)(kt + KVT + r) * DIM + c4, 0, 3);
                __builtin_prefetch(vb + (size_t)(kt + KVT + r) * DIM + c4, 0, 3);
            }
        }
        __syncthreads();

        // scores S = Q K^T  (16 x 64 per wave)
        v8f sc[4];
#pragma unroll
        for (int nb = 0; nb < 4; ++nb) sc[nb] = (v8f)0.0f;
#pragma unroll
        for (int s = 0; s < 4; ++s)
#pragma unroll
            for (int nb = 0; nb < 4; ++nb) {
                v16bf bfrag = load_b_frag<HD>(Ks, nb * 16 + ncol, 32 * s, lane);
                sc[nb] = wmma_bf16(aq[s], bfrag, sc[nb]);
            }

        // online softmax (rows split across half-waves; width-16 reductions)
        float pv[4][8], newm[8];
#pragma unroll
        for (int j = 0; j < 8; ++j) newm[j] = -1e30f;
#pragma unroll
        for (int nb = 0; nb < 4; ++nb)
#pragma unroll
            for (int j = 0; j < 8; ++j) {
                int mg = qtile + wid * 16 + j + halfsel * 8;
                int ng = kt + nb * 16 + ncol;
                float v = sc[nb][j] + mask[(size_t)mg * KVL + ng];
                pv[nb][j] = v;
                newm[j] = fmaxf(newm[j], v);
            }
#pragma unroll
        for (int j = 0; j < 8; ++j) {
#pragma unroll
            for (int d = 1; d < 16; d <<= 1)
                newm[j] = fmaxf(newm[j], __shfl_xor(newm[j], d, 16));
        }
        float scale[8];
#pragma unroll
        for (int j = 0; j < 8; ++j) {
            float mn = fmaxf(mrow[j], newm[j]);
            scale[j] = __expf(mrow[j] - mn);
            mrow[j]  = mn;
        }
        float rsum[8];
#pragma unroll
        for (int j = 0; j < 8; ++j) rsum[j] = 0.0f;
#pragma unroll
        for (int nb = 0; nb < 4; ++nb)
#pragma unroll
            for (int j = 0; j < 8; ++j) {
                float p = __expf(pv[nb][j] - mrow[j]);
                pv[nb][j] = p;
                rsum[j] += p;
            }
#pragma unroll
        for (int j = 0; j < 8; ++j) {
#pragma unroll
            for (int d = 1; d < 16; d <<= 1)
                rsum[j] += __shfl_xor(rsum[j], d, 16);
            lrow[j] = lrow[j] * scale[j] + rsum[j];
        }
#pragma unroll
        for (int t = 0; t < 8; ++t)
#pragma unroll
            for (int j = 0; j < 8; ++j) o[t][j] *= scale[j];

        // stage P (bf16) in this wave's private LDS region, then O += P V
#pragma unroll
        for (int nb = 0; nb < 4; ++nb)
#pragma unroll
            for (int j = 0; j < 8; ++j)
                pw[(j + halfsel * 8) * KVT + nb * 16 + ncol] = (__bf16)pv[nb][j];

#pragma unroll
        for (int s2 = 0; s2 < 2; ++s2) {
            v16bf ap = load_a_frag<KVT>(pw, ncol, 32 * s2, lane);
#pragma unroll
            for (int t = 0; t < 8; ++t) {
                v16bf bv = load_b_frag<KVT>(VTs, t * 16 + ncol, 32 * s2, lane);
                o[t] = wmma_bf16(ap, bv, o[t]);
            }
        }
        __syncthreads();
    }

    // write normalized output in [b, t, h*HD + d] layout (reciprocal, no divides)
    float linv[8];
#pragma unroll
    for (int j = 0; j < 8; ++j) linv[j] = 1.0f / lrow[j];
#pragma unroll
    for (int t = 0; t < 8; ++t)
#pragma unroll
        for (int j = 0; j < 8; ++j) {
            int m = qtile + wid * 16 + j + halfsel * 8;
            int d = h * HD + t * 16 + ncol;
            attn_out[((size_t)(b * TGT + m)) * DIM + d] = o[t][j] * linv[j];
        }
}

// ---------------------------------------------------------------------------
extern "C" void kernel_launch(void* const* d_in, const int* in_sizes, int n_in,
                              void* d_out, int out_size, void* d_ws, size_t ws_size,
                              hipStream_t stream) {
    const float* x    = (const float*)d_in[0];
    const float* kc   = (const float*)d_in[1];
    const float* vc   = (const float*)d_in[2];
    const float* mask = (const float*)d_in[3];
    const float* wq = (const float*)d_in[4];  const float* bq = (const float*)d_in[5];
    const float* wk = (const float*)d_in[6];  const float* bk = (const float*)d_in[7];
    const float* wv = (const float*)d_in[8];  const float* bv = (const float*)d_in[9];
    const float* wo = (const float*)d_in[10]; const float* bo = (const float*)d_in[11];

    float* out   = (float*)d_out;
    float* out_k = out + (size_t)BSZ * TGT * DIM;
    float* out_v = out_k + (size_t)BSZ * KVL * DIM;
    float* q_ws  = (float*)d_ws;
    float* a_ws  = q_ws + (size_t)BSZ * TGT * DIM;

    const int BT = BSZ * TGT;
    const float scale = 0.08838834764831845f;  // 128^-0.5

    // 1) copy unchanged cache rows into the output caches
    copy_old_cache_kernel<<<2048, 256, 0, stream>>>(kc, out_k);
    copy_old_cache_kernel<<<2048, 256, 0, stream>>>(vc, out_v);

    // 2) projections: Q (scaled) -> ws; K,V -> tail rows of output caches
    dim3 ggrid(DIM / 128, BT / 128);
    gemm_bias_wmma_kernel<<<ggrid, 256, 0, stream>>>(x, wq, bq, q_ws, scale,
                                                     DIM, DIM, TGT, 0);
    gemm_bias_wmma_kernel<<<ggrid, 256, 0, stream>>>(x, wk, bk, out_k, 1.0f,
                                                     DIM, DIM, KVL, KVL - TGT);
    gemm_bias_wmma_kernel<<<ggrid, 256, 0, stream>>>(x, wv, bv, out_v, 1.0f,
                                                     DIM, DIM, KVL, KVL - TGT);

    // 3) flash attention over the updated caches
    dim3 agrid(TGT / 128, NH, BSZ);
    flash_attn_wmma_kernel<<<agrid, 256, 0, stream>>>(q_ws, out_k, out_v, mask, a_ws);

    // 4) output projection
    gemm_bias_wmma_kernel<<<ggrid, 256, 0, stream>>>(a_ws, wo, bo, out, 1.0f,
                                                     DIM, DIM, TGT, 0);
}
